// wo2_MoEE_Model_38783554683281
// MI455X (gfx1250) — matrix operand order repaired and checked by reading, not compiled, not run on hardware
//
#include <hip/hip_runtime.h>
#include <hip/hip_bf16.h>

typedef __bf16 bf16;
typedef __attribute__((ext_vector_type(16))) __bf16 v16bf;
typedef __attribute__((ext_vector_type(2)))  __bf16 v2bf;
typedef __attribute__((ext_vector_type(8)))  float  v8f;

// ---- CDNA5 WMMA 16x16x32 bf16 fragment index helpers (ISA 7.12.2) ----
__device__ __forceinline__ int ka_idx(int j, int half) {
  int v = j >> 1, p = j & 1;
  return 16 * (v >> 2) + 8 * half + 2 * (v & 3) + p;
}
__device__ __forceinline__ int kb_idx(int j, int half) {
  int v = j >> 1, p = j & 1;
  return 16 * (v >> 2) + 8 * half + 2 * (v & 3) + p;
}
__device__ __forceinline__ v8f zero8() {
  v8f c;
#pragma unroll
  for (int i = 0; i < 8; ++i) c[i] = 0.f;
  return c;
}
__device__ __forceinline__ v2bf zero2() {
  v2bf z;
  z[0] = (bf16)0.f;
  z[1] = (bf16)0.f;
  return z;
}
__device__ __forceinline__ v8f wmma_bf(v16bf a, v16bf b, v8f c) {
  return __builtin_amdgcn_wmma_f32_16x16x32_bf16(false, a, false, b, (short)0, c,
                                                 false, false);
}
// A fragment via paired (32-bit) loads: K pairs are contiguous in row-major [row][K]
__device__ __forceinline__ v16bf frag_a2(const bf16* rowbase, int koff, int half) {
  union { v16bf v; v2bf h[8]; } u;
#pragma unroll
  for (int vv = 0; vv < 8; ++vv) {
    int k = 16 * (vv >> 2) + 8 * half + 2 * (vv & 3);
    u.h[vv] = *(const v2bf*)(rowbase + koff + k);
  }
  return u.v;
}
// B fragment, scalar gather (for dynamic K-strided operands)
__device__ __forceinline__ v16bf frag_b(const bf16* base, int col, int ld, int koff,
                                        int half) {
  v16bf b;
#pragma unroll
  for (int j = 0; j < 16; ++j) b[j] = base[(koff + kb_idx(j, half)) * ld + col];
  return b;
}
// B fragment from pre-swizzled weights: one aligned 32B vector load
__device__ __forceinline__ v16bf frag_b_sw(const v16bf* base, int s, int ct, int lane) {
  return base[(s * 4 + ct) * 32 + lane];
}

// ---------------- prep kernels ----------------

// Fused conv1(grouped,k=3)+conv2(1x1) -> swizzled Weff (1 kstep, K=24 pad 32), beff
__global__ void k_prep_weff(const float* __restrict__ c1w, const float* __restrict__ c1b,
                            const float* __restrict__ c2w, const float* __restrict__ c2b,
                            bf16* __restrict__ weff, float* __restrict__ beff) {
  int m = blockIdx.x, tid = threadIdx.x;
  for (int e = tid; e < 2048; e += 256) {
    int ct = e >> 9, lane = (e >> 4) & 31, j = e & 15;
    int k = kb_idx(j, lane >> 4);
    int o = ct * 16 + (lane & 15);
    float s = 0.f;
    if (k < 24) {
      for (int i = 0; i < 64; ++i)
        s += c2w[(m * 64 + o) * 64 + i] * c1w[(m * 64 + i) * 24 + k];
    }
    weff[m * 2048 + e] = (bf16)s;
  }
  if (tid < 64) {
    int o = tid;
    float s = c2b[m * 64 + o];
    for (int i = 0; i < 64; ++i) s += c2w[(m * 64 + o) * 64 + i] * c1b[m * 64 + i];
    beff[m * 64 + o] = s;
  }
}

// 64x64 f32 weight -> fragment-swizzled bf16 (2 ksteps x 4 coltiles x 32 lanes x 16)
__global__ void k_swz(const float* __restrict__ src, bf16* __restrict__ dst, int nmat) {
  int idx = blockIdx.x * 256 + threadIdx.x;
  if (idx >= nmat * 4096) return;
  int mat = idx >> 12, e = idx & 4095;
  int s = e >> 11, ct = (e >> 9) & 3, lane = (e >> 4) & 31, j = e & 15;
  int k = s * 32 + kb_idx(j, lane >> 4);
  int col = ct * 16 + (lane & 15);
  dst[idx] = (bf16)src[mat * 4096 + k * 64 + col];
}

// Supports A-frag-swizzled: [t][ks][rt][lane][16]; ks=0: A, ks=1: 2A^2-I
__global__ void k_prep_graph(const float* __restrict__ emb, bf16* __restrict__ sup) {
  int t = blockIdx.x, tid = threadIdx.x;
  __shared__ float Af[32][32];
  if (tid < 32) {
    int n = tid;
    float e0 = emb[(t * 32 + n) * 2], e1 = emb[(t * 32 + n) * 2 + 1];
    float row[32];
    float mx = -1e30f;
    for (int mm = 0; mm < 32; ++mm) {
      float d = e0 * emb[(t * 32 + mm) * 2] + e1 * emb[(t * 32 + mm) * 2 + 1];
      d = fmaxf(d, 0.f);
      row[mm] = d;
      mx = fmaxf(mx, d);
    }
    float s = 0.f;
    for (int mm = 0; mm < 32; ++mm) { row[mm] = __expf(row[mm] - mx); s += row[mm]; }
    float inv = 1.f / s;
    for (int mm = 0; mm < 32; ++mm) Af[n][mm] = row[mm] * inv;
  }
  __syncthreads();
  for (int e = tid; e < 2048; e += 256) {
    int ks = e >> 10, lane = (e >> 4) & 31, j = e & 15;
    int n = (((e >> 9) & 1) << 4) + (lane & 15);
    int mm = ka_idx(j, lane >> 4);
    float val;
    if (ks == 0) {
      val = Af[n][mm];
    } else {
      float a2 = 0.f;
      for (int j2 = 0; j2 < 32; ++j2) a2 += Af[n][j2] * Af[j2][mm];
      val = 2.f * a2 - (n == mm ? 1.f : 0.f);
    }
    sup[t * 2048 + e] = (bf16)val;
  }
}

// DAGCN weights: swizzled [t*32+n][6 ksteps][4 ct][32 lane][16]; bias_t[t,n,o]
__global__ void k_prep_wpool(const float* __restrict__ emb, const float* __restrict__ wpool,
                             const float* __restrict__ bpool, bf16* __restrict__ wbf,
                             float* __restrict__ biasT) {
  int tn = blockIdx.x, tid = threadIdx.x;
  float e0 = emb[tn * 2], e1 = emb[tn * 2 + 1];
  for (int idx = tid; idx < 12288; idx += 256) {
    int s = idx >> 11, ct = (idx >> 9) & 3, lane = (idx >> 4) & 31, j = idx & 15;
    int k = s * 32 + kb_idx(j, lane >> 4);  // 0..191 = kcheb*64 + i
    int o = ct * 16 + (lane & 15);
    int kc = k >> 6, i = k & 63;
    float v = e0 * wpool[((0 * 3 + kc) * 64 + i) * 64 + o] +
              e1 * wpool[((1 * 3 + kc) * 64 + i) * 64 + o];
    wbf[(size_t)tn * 12288 + idx] = (bf16)v;
  }
  if (tid < 64) biasT[tn * 64 + tid] = e0 * bpool[tid] + e1 * bpool[64 + tid];
}

// ---------------- fused conv embedding GEMM: x[b,m,t,o] ----------------
__global__ void k_embed(const float* __restrict__ inx, const bf16* __restrict__ weff,
                        const float* __restrict__ beff, float* __restrict__ xf) {
  int tid = threadIdx.x;
  int w = tid >> 5, lane = tid & 31, half = lane >> 4, l16 = lane & 15;
  int m = blockIdx.y;
  int rowbase = blockIdx.x * 128 + w * 16;  // rows = b*30+t
  int row = rowbase + l16;
  int bb = row / 30, tt = row - bb * 30;
  v16bf a;
#pragma unroll
  for (int j = 0; j < 16; ++j) {
    int kk = ka_idx(j, half);
    float v = 0.f;
    if (kk < 24) v = inx[bb * 8192 + (tt + kk % 3) * 256 + m * 8 + (kk / 3)];
    a[j] = (bf16)v;
  }
  const v16bf* wb = (const v16bf*)weff + m * 128;
  for (int ct = 0; ct < 4; ++ct) {
    v8f c = wmma_bf(a, frag_b_sw(wb, 0, ct, lane), zero8());
    int o = ct * 16 + l16;
    float bias = beff[m * 64 + o];
#pragma unroll
    for (int r = 0; r < 8; ++r) {
      int row2 = rowbase + r + 8 * half;
      int b2 = row2 / 30, t2 = row2 - b2 * 30;
      xf[((size_t)(b2 * 32 + m) * 30 + t2) * 64 + o] = c[r] + bias;
    }
  }
}

// -------- fused RoPE-MHA per (b,m): qkv->rope->attn->wo->+x->LN --------
__global__ void k_att(const float* __restrict__ xf, const bf16* __restrict__ wq,
                      const bf16* __restrict__ wk, const bf16* __restrict__ wv,
                      const bf16* __restrict__ wo, const float* __restrict__ bq,
                      const float* __restrict__ bk, const float* __restrict__ bv,
                      const float* __restrict__ lng, const float* __restrict__ lnb,
                      float* __restrict__ hf) {
  int m = blockIdx.x, b = blockIdx.y, tid = threadIdx.x;
  int w = tid >> 5, lane = tid & 31, half = lane >> 4, l16 = lane & 15;

  __shared__ float xs[1920];
  __shared__ float hb[1920];
  __shared__ __align__(32) bf16 xbf[2048];
  __shared__ __align__(32) bf16 qkvbf[3 * 2048];
  __shared__ float red[512];
  __shared__ __align__(32) char regA[24576];  // phase1: qkv f32; phase2: P + attn
  float* qkvf = (float*)regA;                 // [3][32][64]
  bf16* pbf = (bf16*)regA;                    // [8][32][32]
  bf16* attbf = (bf16*)(regA + 16384);        // [32][64]

  const float* xsrc = xf + (size_t)((b * 32 + m) * 30) * 64;
  for (int i = tid; i < 2048; i += 256) {
    float v = (i < 1920) ? xsrc[i] : 0.f;
    if (i < 1920) xs[i] = v;
    xbf[i] = (bf16)v;
  }
  __syncthreads();

  int rt = w >> 2, ct = w & 3;
  {  // q,k,v projections: wave owns its (rt,ct) tile for all three
    v16bf a0 = frag_a2(xbf + (rt * 16 + l16) * 64, 0, half);
    v16bf a1 = frag_a2(xbf + (rt * 16 + l16) * 64, 32, half);
    const v16bf* wmats[3] = {(const v16bf*)wq + m * 256, (const v16bf*)wk + m * 256,
                             (const v16bf*)wv + m * 256};
    const float* bvecs[3] = {bq + m * 64, bk + m * 64, bv + m * 64};
    for (int p = 0; p < 3; ++p) {
      v8f c = wmma_bf(a0, frag_b_sw(wmats[p], 0, ct, lane), zero8());
      c = wmma_bf(a1, frag_b_sw(wmats[p], 1, ct, lane), c);
      int o = ct * 16 + l16;
      float bias = bvecs[p][o];
#pragma unroll
      for (int r = 0; r < 8; ++r)
        qkvf[p * 2048 + (rt * 16 + r + 8 * half) * 64 + o] = c[r] + bias;
    }
  }
  __syncthreads();

  // RoPE on q,k + bf16 convert; rows 30,31 zeroed (K padding safety)
  for (int idx = tid; idx < 6144; idx += 256) {
    int p = idx >> 11, rem = idx & 2047, trow = rem >> 6, e = rem & 63;
    float out = 0.f;
    if (trow < 30) {
      if (p == 2) {
        out = qkvf[idx];
      } else {
        int eh = e & 7, f = eh >> 1;
        float xe = qkvf[p * 2048 + trow * 64 + (e & ~1)];
        float xo = qkvf[p * 2048 + trow * 64 + (e | 1)];
        float freq = __powf(10000.f, -(float)f * 0.25f);
        float ang = (float)trow * freq;
        float cs = __cosf(ang), sn = __sinf(ang);
        out = (eh & 1) ? (xe * sn + xo * cs) : (xe * cs - xo * sn);
      }
    }
    qkvbf[idx] = (bf16)out;
  }
  __syncthreads();

  {  // attention: wave w = head h; S^T so softmax reduces per-lane (+xor16)
    int h = w;
    const bf16* qb2 = qkvbf;
    const bf16* kb2 = qkvbf + 2048;
    const bf16* vb2 = qkvbf + 4096;
    v16bf ak[2], bq2[2];
#pragma unroll
    for (int ki = 0; ki < 2; ++ki) {  // A = K-matrix rows, contiguous head-dim pairs
      union { v16bf v; v2bf h2[8]; } u;
#pragma unroll
      for (int vv = 0; vv < 8; ++vv) {
        int ke = 16 * (vv >> 2) + 8 * half + 2 * (vv & 3);
        u.h2[vv] = (ke < 8) ? *(const v2bf*)(kb2 + (ki * 16 + l16) * 64 + h * 8 + ke)
                            : zero2();
      }
      ak[ki] = u.v;
    }
#pragma unroll
    for (int qi = 0; qi < 2; ++qi) {  // B = Q^T, contiguous head-dim pairs
      union { v16bf v; v2bf h2[8]; } u;
#pragma unroll
      for (int vv = 0; vv < 8; ++vv) {
        int ke = 16 * (vv >> 2) + 8 * half + 2 * (vv & 3);
        u.h2[vv] = (ke < 8) ? *(const v2bf*)(qb2 + (qi * 16 + l16) * 64 + h * 8 + ke)
                            : zero2();
      }
      bq2[qi] = u.v;
    }
    v8f sc[2][2];
    for (int ki = 0; ki < 2; ++ki)
      for (int qi = 0; qi < 2; ++qi) sc[ki][qi] = wmma_bf(ak[ki], bq2[qi], zero8());

    const float scale = 0.125f;  // 1/sqrt(D=64)
    for (int qi = 0; qi < 2; ++qi) {
      float mx = -1e30f;
#pragma unroll
      for (int ki = 0; ki < 2; ++ki)
#pragma unroll
        for (int r = 0; r < 8; ++r) {
          int kk = ki * 16 + r + 8 * half;
          if (kk < 30) mx = fmaxf(mx, sc[ki][qi][r] * scale);
        }
      mx = fmaxf(mx, __shfl_xor(mx, 16, 32));
      float sum = 0.f;
      float pv[2][8];
#pragma unroll
      for (int ki = 0; ki < 2; ++ki)
#pragma unroll
        for (int r = 0; r < 8; ++r) {
          int kk = ki * 16 + r + 8 * half;
          float e2 = (kk < 30) ? __expf(sc[ki][qi][r] * scale - mx) : 0.f;
          pv[ki][r] = e2;
          sum += e2;
        }
      sum += __shfl_xor(sum, 16, 32);
      float inv = 1.f / sum;
#pragma unroll
      for (int ki = 0; ki < 2; ++ki)
#pragma unroll
        for (int r = 0; r < 8; ++r) {
          int kk = ki * 16 + r + 8 * half;
          pbf[h * 1024 + kk * 32 + qi * 16 + l16] = (bf16)(pv[ki][r] * inv);
        }
    }
    // o = P @ V (K = 30 padded to 32; P pad rows are 0)
    for (int qi = 0; qi < 2; ++qi) {
      v16bf a;
#pragma unroll
      for (int j = 0; j < 16; ++j)
        a[j] = pbf[h * 1024 + ka_idx(j, half) * 32 + qi * 16 + l16];
      v16bf bb;
#pragma unroll
      for (int j = 0; j < 16; ++j)
        bb[j] = (l16 < 8) ? vb2[kb_idx(j, half) * 64 + h * 8 + l16] : (bf16)0.f;
      v8f c = wmma_bf(a, bb, zero8());
      if (l16 < 8) {
#pragma unroll
        for (int r = 0; r < 8; ++r)
          attbf[(qi * 16 + r + 8 * half) * 64 + h * 8 + l16] = (bf16)c[r];
      }
    }
  }
  __syncthreads();

  {  // out projection + residual
    v16bf a0 = frag_a2(attbf + (rt * 16 + l16) * 64, 0, half);
    v16bf a1 = frag_a2(attbf + (rt * 16 + l16) * 64, 32, half);
    const v16bf* wom = (const v16bf*)wo + m * 256;
    v8f c = wmma_bf(a0, frag_b_sw(wom, 0, ct, lane), zero8());
    c = wmma_bf(a1, frag_b_sw(wom, 1, ct, lane), c);
#pragma unroll
    for (int r = 0; r < 8; ++r) {
      int row = rt * 16 + r + 8 * half;
      if (row < 30) {
        int o = ct * 16 + l16;
        hb[row * 64 + o] = c[r] + xs[row * 64 + o];
      }
    }
  }
  __syncthreads();

  // LayerNorm over (T,D)
  float s = 0.f, s2 = 0.f;
  for (int i = tid; i < 1920; i += 256) { float v = hb[i]; s += v; s2 += v * v; }
  red[tid] = s; red[256 + tid] = s2;
  __syncthreads();
  for (int off = 128; off > 0; off >>= 1) {
    if (tid < off) { red[tid] += red[tid + off]; red[256 + tid] += red[256 + tid + off]; }
    __syncthreads();
  }
  float mu = red[0] * (1.f / 1920.f);
  float var = red[256] * (1.f / 1920.f) - mu * mu;
  float rinv = rsqrtf(var + 1e-5f);
  float* hdst = hf + (size_t)((b * 32 + m) * 30) * 64;
  for (int i = tid; i < 1920; i += 256)
    hdst[i] = (hb[i] - mu) * rinv * lng[i] + lnb[i];
}

// ------- per-node FFN + residual + LN(M,T,D); writes transposed bf16 -------
__global__ void k_iffn(const float* __restrict__ hf, const bf16* __restrict__ w1,
                       const bf16* __restrict__ w2, const float* __restrict__ b1,
                       const float* __restrict__ b2, const float* __restrict__ lng,
                       const float* __restrict__ lnb, bf16* __restrict__ htbf) {
  int m = blockIdx.x, b = blockIdx.y, tid = threadIdx.x;
  int w = tid >> 5, lane = tid & 31, half = lane >> 4, l16 = lane & 15;
  __shared__ float hs[1920];
  __shared__ __align__(32) bf16 hbf[2048];
  __shared__ __align__(32) bf16 f1bf[2048];
  __shared__ float fb[1920];
  __shared__ float red[512];
  const float* src = hf + (size_t)((b * 32 + m) * 30) * 64;
  for (int i = tid; i < 2048; i += 256) {
    float v = (i < 1920) ? src[i] : 0.f;
    if (i < 1920) hs[i] = v;
    hbf[i] = (bf16)v;
  }
  __syncthreads();
  int rt = w >> 2, ct = w & 3;
  int o = ct * 16 + l16;
  {
    v16bf a0 = frag_a2(hbf + (rt * 16 + l16) * 64, 0, half);
    v16bf a1 = frag_a2(hbf + (rt * 16 + l16) * 64, 32, half);
    const v16bf* wm = (const v16bf*)w1 + m * 256;
    v8f c = wmma_bf(a0, frag_b_sw(wm, 0, ct, lane), zero8());
    c = wmma_bf(a1, frag_b_sw(wm, 1, ct, lane), c);
    float bias = b1[m * 64 + o];
#pragma unroll
    for (int r = 0; r < 8; ++r)
      f1bf[(rt * 16 + r + 8 * half) * 64 + o] = (bf16)fmaxf(c[r] + bias, 0.f);
  }
  __syncthreads();
  {
    v16bf a0 = frag_a2(f1bf + (rt * 16 + l16) * 64, 0, half);
    v16bf a1 = frag_a2(f1bf + (rt * 16 + l16) * 64, 32, half);
    const v16bf* wm = (const v16bf*)w2 + m * 256;
    v8f c = wmma_bf(a0, frag_b_sw(wm, 0, ct, lane), zero8());
    c = wmma_bf(a1, frag_b_sw(wm, 1, ct, lane), c);
    float bias = b2[m * 64 + o];
#pragma unroll
    for (int r = 0; r < 8; ++r) {
      int row = rt * 16 + r + 8 * half;
      if (row < 30) fb[row * 64 + o] = c[r] + bias + hs[row * 64 + o];
    }
  }
  __syncthreads();
  float s = 0.f, s2 = 0.f;
  for (int i = tid; i < 1920; i += 256) { float v = fb[i]; s += v; s2 += v * v; }
  red[tid] = s; red[256 + tid] = s2;
  __syncthreads();
  for (int off = 128; off > 0; off >>= 1) {
    if (tid < off) { red[tid] += red[tid + off]; red[256 + tid] += red[256 + tid + off]; }
    __syncthreads();
  }
  float mu = red[0] * (1.f / 1920.f);
  float var = red[256] * (1.f / 1920.f) - mu * mu;
  float rinv = rsqrtf(var + 1e-6f);
  for (int i = tid; i < 1920; i += 256) {
    int t = i >> 6, d = i & 63;
    float v = (fb[i] - mu) * rinv * lng[m * 1920 + i] + lnb[m * 1920 + i];
    htbf[((size_t)(b * 30 + t) * 32 + m) * 64 + d] = (bf16)v;
  }
}

// ------------- DAGCN: xg_k = supports_k @ H_t per (b,t) -------------
__global__ void k_xg(const bf16* __restrict__ htbf, const bf16* __restrict__ sup,
                     bf16* __restrict__ xgbf) {
  int t = blockIdx.x, b = blockIdx.y, tid = threadIdx.x;
  int w = tid >> 5, lane = tid & 31, half = lane >> 4, l16 = lane & 15;
  __shared__ __align__(16) bf16 hm[2048];
  {
    const uint4* s128 = (const uint4*)(htbf + (size_t)(b * 30 + t) * 2048);
    uint4* d128 = (uint4*)hm;
    if (tid < 256) d128[tid] = s128[tid];
  }
  __syncthreads();
  const v16bf* supf = (const v16bf*)sup;
  for (int jid = w; jid < 16; jid += 8) {
    int ks = jid >> 3, tile = jid & 7, rt = tile >> 2, ct = tile & 3;
    v16bf a = supf[((t * 2 + ks) * 2 + rt) * 32 + lane];
    v16bf bb = frag_b(hm, ct * 16 + l16, 64, 0, half);
    v8f c = wmma_bf(a, bb, zero8());
#pragma unroll
    for (int r = 0; r < 8; ++r) {
      int n2 = rt * 16 + r + 8 * half;
      xgbf[((size_t)((t * 2 + ks) * 32 + n2) * 512 + b) * 64 + ct * 16 + l16] =
          (bf16)c[r];
    }
  }
}

// ----- DAGCN contraction: rows=b, K=192 (k,i), N=64; +bias +x -> z -----
__global__ void k_g(const bf16* __restrict__ htbf, const bf16* __restrict__ xgbf,
                    const bf16* __restrict__ wbf, const float* __restrict__ biasT,
                    const float* __restrict__ xf, float* __restrict__ zf) {
  int bt = blockIdx.x, n = blockIdx.y, t = blockIdx.z, tid = threadIdx.x;
  int w = tid >> 5, lane = tid & 31, half = lane >> 4, l16 = lane & 15;
  __shared__ __align__(32) bf16 Wl[12288];  // swizzled [6][4][32][16]
  {
    const uint4* s128 = (const uint4*)(wbf + (size_t)(t * 32 + n) * 12288);
    uint4* d128 = (uint4*)Wl;
    for (int i = tid; i < 1536; i += 256) d128[i] = s128[i];
  }
  __syncthreads();
  int rbase = bt * 128 + w * 16;
  int brow = rbase + l16;
  const v16bf* wlf = (const v16bf*)Wl;
  v8f acc[4];
  for (int ct = 0; ct < 4; ++ct) acc[ct] = zero8();
  for (int s = 0; s < 6; ++s) {
    int k = s >> 1;
    const bf16* rowp = (k == 0)
        ? htbf + (size_t)((brow * 30 + t) * 32 + n) * 64
        : xgbf + ((size_t)((t * 2 + (k - 1)) * 32 + n) * 512 + brow) * 64;
    v16bf a = frag_a2(rowp, (s & 1) * 32, half);
    for (int ct = 0; ct < 4; ++ct)
      acc[ct] = wmma_bf(a, wlf[(s * 4 + ct) * 32 + lane], acc[ct]);
  }
  for (int ct = 0; ct < 4; ++ct) {
    int o = ct * 16 + l16;
    float bias = biasT[(t * 32 + n) * 64 + o];
#pragma unroll
    for (int r = 0; r < 8; ++r) {
      int b2 = rbase + r + 8 * half;
      size_t zi = ((size_t)(b2 * 32 + n) * 30 + t) * 64 + o;
      zf[zi] = acc[ct][r] + bias + xf[zi];
    }
  }
}

// ------- shared FFN + LN + time/feature heads per (b,m) -------
__global__ void k_final(const float* __restrict__ zf, const bf16* __restrict__ w1,
                        const bf16* __restrict__ w2, const float* __restrict__ b1,
                        const float* __restrict__ b2, const float* __restrict__ lng,
                        const float* __restrict__ lnb, const float* __restrict__ timew,
                        const float* __restrict__ timeb, const float* __restrict__ featw,
                        float* __restrict__ partial) {
  int m = blockIdx.x, b = blockIdx.y, tid = threadIdx.x;
  int w = tid >> 5, lane = tid & 31, half = lane >> 4, l16 = lane & 15;
  __shared__ float hs[1920];
  __shared__ __align__(32) bf16 hbf[2048];
  __shared__ __align__(32) bf16 f1bf[2048];
  __shared__ float fb[1920];
  __shared__ float red[512];
  const float* src = zf + (size_t)((b * 32 + m) * 30) * 64;
  for (int i = tid; i < 2048; i += 256) {
    float v = (i < 1920) ? src[i] : 0.f;
    if (i < 1920) hs[i] = v;
    hbf[i] = (bf16)v;
  }
  __syncthreads();
  int rt = w >> 2, ct = w & 3;
  int o = ct * 16 + l16;
  {
    v16bf a0 = frag_a2(hbf + (rt * 16 + l16) * 64, 0, half);
    v16bf a1 = frag_a2(hbf + (rt * 16 + l16) * 64, 32, half);
    const v16bf* wm = (const v16bf*)w1;
    v8f c = wmma_bf(a0, frag_b_sw(wm, 0, ct, lane), zero8());
    c = wmma_bf(a1, frag_b_sw(wm, 1, ct, lane), c);
    float bias = b1[o];
#pragma unroll
    for (int r = 0; r < 8; ++r)
      f1bf[(rt * 16 + r + 8 * half) * 64 + o] = (bf16)fmaxf(c[r] + bias, 0.f);
  }
  __syncthreads();
  {
    v16bf a0 = frag_a2(f1bf + (rt * 16 + l16) * 64, 0, half);
    v16bf a1 = frag_a2(f1bf + (rt * 16 + l16) * 64, 32, half);
    const v16bf* wm = (const v16bf*)w2;
    v8f c = wmma_bf(a0, frag_b_sw(wm, 0, ct, lane), zero8());
    c = wmma_bf(a1, frag_b_sw(wm, 1, ct, lane), c);
    float bias = b2[o];
#pragma unroll
    for (int r = 0; r < 8; ++r) {
      int row = rt * 16 + r + 8 * half;
      if (row < 30) fb[row * 64 + o] = c[r] + bias + hs[row * 64 + o];
    }
  }
  __syncthreads();
  float s = 0.f, s2 = 0.f;
  for (int i = tid; i < 1920; i += 256) { float v = fb[i]; s += v; s2 += v * v; }
  red[tid] = s; red[256 + tid] = s2;
  __syncthreads();
  for (int off = 128; off > 0; off >>= 1) {
    if (tid < off) { red[tid] += red[tid + off]; red[256 + tid] += red[256 + tid + off]; }
    __syncthreads();
  }
  float mu = red[0] * (1.f / 1920.f);
  float var = red[256] * (1.f / 1920.f) - mu * mu;
  float rinv = rsqrtf(var + 1e-6f);
  for (int i = tid; i < 1920; i += 256)
    hs[i] = (fb[i] - mu) * rinv * lng[i] + lnb[i];  // reuse hs for normalized y
  __syncthreads();
  if (tid < 64) {
    float rd = 0.f;
    for (int t = 0; t < 30; ++t) rd += hs[t * 64 + tid] * timew[t];
    red[tid] = (rd + timeb[0]) * featw[m * 64 + tid];
  }
  __syncthreads();
  if (tid == 0) {
    float sfin = 0.f;
    for (int i = 0; i < 64; ++i) sfin += red[i];
    partial[b * 32 + m] = sfin;
  }
}

__global__ void k_reduce(const float* __restrict__ partial, const float* __restrict__ featb,
                         float* __restrict__ out) {
  int b = blockIdx.x * blockDim.x + threadIdx.x;
  if (b < 512) {
    float s = featb[0];
    for (int m = 0; m < 32; ++m) s += partial[b * 32 + m];
    out[b] = s;
  }
}

// ---------------- host launcher ----------------
extern "C" void kernel_launch(void* const* d_in, const int* in_sizes, int n_in,
                              void* d_out, int out_size, void* d_ws, size_t ws_size,
                              hipStream_t stream) {
  const float* inx = (const float*)d_in[0];
  const float* c1w = (const float*)d_in[1];
  const float* c1b = (const float*)d_in[2];
  const float* c2w = (const float*)d_in[3];
  const float* c2b = (const float*)d_in[4];
  const float* wq = (const float*)d_in[5];
  const float* bq = (const float*)d_in[6];
  const float* wk = (const float*)d_in[7];
  const float* bk = (const float*)d_in[8];
  const float* wv = (const float*)d_in[9];
  const float* bv = (const float*)d_in[10];
  const float* wo = (const float*)d_in[11];
  const float* lng = (const float*)d_in[12];
  const float* lnb = (const float*)d_in[13];
  const float* if1 = (const float*)d_in[14];
  const float* ib1 = (const float*)d_in[15];
  const float* if2 = (const float*)d_in[16];
  const float* ib2 = (const float*)d_in[17];
  const float* ilng = (const float*)d_in[18];
  const float* ilnb = (const float*)d_in[19];
  const float* emb = (const float*)d_in[20];
  const float* wpool = (const float*)d_in[21];
  const float* bpool = (const float*)d_in[22];
  const float* f1 = (const float*)d_in[23];
  const float* fb1 = (const float*)d_in[24];
  const float* f2 = (const float*)d_in[25];
  const float* fb2 = (const float*)d_in[26];
  const float* flng = (const float*)d_in[27];
  const float* flnb = (const float*)d_in[28];
  const float* timew = (const float*)d_in[29];
  const float* timeb = (const float*)d_in[30];
  const float* featw = (const float*)d_in[31];
  const float* featb = (const float*)d_in[32];

  char* ws = (char*)d_ws;
  size_t off = 0;
  auto alloc = [&](size_t bytes) -> void* {
    void* p = ws + off;
    off += (bytes + 255) & ~(size_t)255;
    return p;
  };
  bf16* weff   = (bf16*)alloc((size_t)32 * 2048 * 2);
  float* beff  = (float*)alloc((size_t)32 * 64 * 4);
  bf16* wq_bf  = (bf16*)alloc((size_t)32 * 4096 * 2);
  bf16* wk_bf  = (bf16*)alloc((size_t)32 * 4096 * 2);
  bf16* wv_bf  = (bf16*)alloc((size_t)32 * 4096 * 2);
  bf16* wo_bf  = (bf16*)alloc((size_t)32 * 4096 * 2);
  bf16* if1_bf = (bf16*)alloc((size_t)32 * 4096 * 2);
  bf16* if2_bf = (bf16*)alloc((size_t)32 * 4096 * 2);
  bf16* f1_bf  = (bf16*)alloc((size_t)4096 * 2);
  bf16* f2_bf  = (bf16*)alloc((size_t)4096 * 2);
  bf16* sup_bf = (bf16*)alloc((size_t)30 * 2048 * 2);
  bf16* wp_bf  = (bf16*)alloc((size_t)960 * 12288 * 2);
  float* biasT = (float*)alloc((size_t)960 * 64 * 4);
  float* x_f   = (float*)alloc((size_t)512 * 32 * 30 * 64 * 4);
  float* h_f   = (float*)alloc((size_t)512 * 32 * 30 * 64 * 4);  // reused as z
  bf16* ht_bf  = (bf16*)alloc((size_t)512 * 30 * 32 * 64 * 2);
  bf16* xg_bf  = (bf16*)alloc((size_t)30 * 2 * 32 * 512 * 64 * 2);
  float* part  = (float*)alloc((size_t)512 * 32 * 4);

  k_prep_weff<<<32, 256, 0, stream>>>(c1w, c1b, c2w, c2b, weff, beff);
  k_swz<<<512, 256, 0, stream>>>(wq, wq_bf, 32);
  k_swz<<<512, 256, 0, stream>>>(wk, wk_bf, 32);
  k_swz<<<512, 256, 0, stream>>>(wv, wv_bf, 32);
  k_swz<<<512, 256, 0, stream>>>(wo, wo_bf, 32);
  k_swz<<<512, 256, 0, stream>>>(if1, if1_bf, 32);
  k_swz<<<512, 256, 0, stream>>>(if2, if2_bf, 32);
  k_swz<<<16, 256, 0, stream>>>(f1, f1_bf, 1);
  k_swz<<<16, 256, 0, stream>>>(f2, f2_bf, 1);
  k_prep_graph<<<30, 256, 0, stream>>>(emb, sup_bf);
  k_prep_wpool<<<960, 256, 0, stream>>>(emb, wpool, bpool, wp_bf, biasT);

  k_embed<<<dim3(120, 32), 256, 0, stream>>>(inx, weff, beff, x_f);
  k_att<<<dim3(32, 512), 256, 0, stream>>>(x_f, wq_bf, wk_bf, wv_bf, wo_bf, bq, bk, bv,
                                           lng, lnb, h_f);
  k_iffn<<<dim3(32, 512), 256, 0, stream>>>(h_f, if1_bf, if2_bf, ib1, ib2, ilng, ilnb,
                                            ht_bf);
  k_xg<<<dim3(30, 512), 256, 0, stream>>>(ht_bf, sup_bf, xg_bf);
  k_g<<<dim3(4, 32, 30), 256, 0, stream>>>(ht_bf, xg_bf, wp_bf, biasT, x_f, h_f);
  k_final<<<dim3(32, 512), 256, 0, stream>>>(h_f, f1_bf, f2_bf, fb1, fb2, flng, flnb,
                                             timew, timeb, featw, part);
  k_reduce<<<2, 256, 0, stream>>>(part, featb, (float*)d_out);
}
